// SGFE_2001454760191
// MI455X (gfx1250) — compile-verified, matrix operand order checked by reading, not run on hardware
//
#include <hip/hip_runtime.h>
#include <math.h>

// ---------------------------------------------------------------------------
// Types for CDNA5 WMMA / TDM (gfx1250, wave32)
// ---------------------------------------------------------------------------
typedef __attribute__((ext_vector_type(16))) __bf16 v16bf;
typedef __attribute__((ext_vector_type(8)))  float  v8f;
typedef __attribute__((ext_vector_type(4)))  unsigned int u32x4;
typedef __attribute__((ext_vector_type(8)))  int          i32x8;
typedef __attribute__((ext_vector_type(4)))  int          i32x4;

static constexpr int CFEAT   = 64;
static constexpr int TM      = 128;            // points per MLP block
static constexpr int APITCH  = 72;             // bf16 tile row pitch (pad vs 64)
static constexpr int FPITCH  = 68;             // f32 tile row pitch (TDM pad: 64+4 dwords)
static constexpr unsigned HBITS  = 20;
static constexpr unsigned HSIZE  = 1u << HBITS;
static constexpr unsigned HMASK  = HSIZE - 1u;
static constexpr unsigned KEMPTY = 0xFFFFFFFFu;

__device__ __forceinline__ unsigned hash_key(unsigned k) {
    k *= 2654435761u;
    k ^= k >> 15;
    return k & HMASK;
}

// ---------------------------------------------------------------------------
// Phase 1: hash insert (voxel key -> slot) + per-voxel point count
// ---------------------------------------------------------------------------
__global__ __launch_bounds__(256) void k_insert(const int* __restrict__ coords,
                                                unsigned* __restrict__ keys,
                                                float* __restrict__ cnt,
                                                unsigned* __restrict__ slots,
                                                int N, int scale) {
    int p = blockIdx.x * blockDim.x + threadIdx.x;
    if (p >= N) return;
    int x = coords[p * 4 + 0], y = coords[p * 4 + 1];
    int z = coords[p * 4 + 2], b = coords[p * 4 + 3];
    unsigned key = ((unsigned)(b * 512 + x / scale) * 512u + (unsigned)(y / scale)) * 512u
                   + (unsigned)(z / scale);
    unsigned s = hash_key(key);
    for (;;) {
        unsigned old = atomicCAS(&keys[s], KEMPTY, key);
        if (old == KEMPTY || old == key) break;
        s = (s + 1u) & HMASK;
    }
    slots[p] = s;
    atomicAdd(&cnt[s], 1.0f);
}

// ---------------------------------------------------------------------------
// Phase 2: scatter-add features into voxel sums (segment_sum equivalent)
// ---------------------------------------------------------------------------
__global__ __launch_bounds__(256) void k_scatter(const float* __restrict__ feats,
                                                 const unsigned* __restrict__ slots,
                                                 float* __restrict__ hsum, int N) {
    long long t = (long long)blockIdx.x * blockDim.x + threadIdx.x;
    if (t >= (long long)N * CFEAT) return;
    int p = (int)(t >> 6), c = (int)(t & 63);
    atomicAdd(&hsum[(size_t)slots[p] * CFEAT + c], feats[(size_t)p * CFEAT + c]);
}

// ---------------------------------------------------------------------------
// Phase 3: nearest-existing-corner selection (searchsorted + argmax(w))
// ---------------------------------------------------------------------------
__global__ __launch_bounds__(256) void k_corner(const int* __restrict__ coords,
                                                const unsigned* __restrict__ keys,
                                                const unsigned* __restrict__ slots,
                                                unsigned* __restrict__ best,
                                                int N, int scale) {
    int p = blockIdx.x * blockDim.x + threadIdx.x;
    if (p >= N) return;
    int x = coords[p * 4 + 0], y = coords[p * 4 + 1];
    int z = coords[p * 4 + 2], b = coords[p * 4 + 3];
    int vx = x / scale, vy = y / scale, vz = z / scale;
    float inv = 1.0f / (float)scale;
    float fx = (float)(x - vx * scale) * inv;
    float fy = (float)(y - vy * scale) * inv;
    float fz = (float)(z - vz * scale) * inv;
    float bw = -1.0f;
    unsigned bslot = slots[p];
    #pragma unroll
    for (int j = 0; j < 8; ++j) {
        int bx = (j >> 2) & 1, by = (j >> 1) & 1, bz = j & 1;
        float w = (bx ? fx : 1.0f - fx) * (by ? fy : 1.0f - fy) * (bz ? fz : 1.0f - fz);
        unsigned key = ((unsigned)(b * 512 + vx + bx) * 512u + (unsigned)(vy + by)) * 512u
                       + (unsigned)(vz + bz);
        unsigned s = hash_key(key);
        bool found = false;
        for (;;) {
            unsigned k = keys[s];
            if (k == key) { found = true; break; }
            if (k == KEMPTY) break;
            s = (s + 1u) & HMASK;
        }
        if (found && w > bw) { bw = w; bslot = s; }   // first-max wins (strict >)
    }
    best[p] = bslot;
}

// ---------------------------------------------------------------------------
// WMMA fragment loader: row-major 16-bit tile, ISA 16x32 A/B layout.
// lane 0..15 -> row m, lane>>4 selects K-half; two contiguous 8-elem chunks.
// ---------------------------------------------------------------------------
__device__ __forceinline__ v16bf load_frag(const __bf16* base, int stride,
                                           int mbase, int kbase, int lane) {
    const int m  = mbase + (lane & 15);
    const int kh = (lane >> 4) << 3;            // 0 or 8
    const __bf16* r = base + m * stride + kbase + kh;
    v16bf v;
    #pragma unroll
    for (int e = 0; e < 8; ++e) { v[e] = r[e]; v[e + 8] = r[e + 16]; }
    return v;
}

// ---------------------------------------------------------------------------
// Phase 4: fused gather-residual + 2-layer MLP via v_wmma_f32_16x16x32_bf16.
// The feats tile (128 x 64 f32) is DMA'd global->LDS by the Tensor Data Mover:
//   - tensor_dim1 = N  => OOB tail rows are zero-filled by hardware
//   - pad_interval=5 (64 dwords), pad_amount=3 (4 dwords) => LDS pitch 68 dwords
// Block: 256 threads = 8 waves; 128 points; wave w owns rows [16w,16w+16).
// ---------------------------------------------------------------------------
__global__ __launch_bounds__(256) void k_mlp(const float* __restrict__ feats,
                                             const float* __restrict__ hsum,
                                             const float* __restrict__ hcnt,
                                             const unsigned* __restrict__ best,
                                             const float* __restrict__ W1,
                                             const float* __restrict__ b1,
                                             const float* __restrict__ W2,
                                             const float* __restrict__ b2,
                                             float* __restrict__ msOut, int N) {
    __shared__ __bf16 sW1t[CFEAT * CFEAT];      // transposed: [n][k]
    __shared__ __bf16 sW2t[CFEAT * CFEAT];
    __shared__ float  sB1[CFEAT], sB2[CFEAT];
    __shared__ __bf16 sA[TM * APITCH];          // residual tile (bf16)
    __shared__ float  sF[TM * FPITCH];          // feats tile (f32), TDM destination
    __shared__ __bf16 sH[TM * APITCH];          // hidden tile (bf16)
    __shared__ unsigned sSlot[TM];
    __shared__ float    sInvC[TM];

    const int tid = threadIdx.x;
    const int p0  = blockIdx.x * TM;

    // ---- TDM: DMA the feats tile into sF (wave 0 issues; EXEC is ignored) ----
    if (tid == 0) {
        unsigned long long gaddr = (unsigned long long)(const void*)(feats + (size_t)p0 * CFEAT);
        unsigned ldsAddr = (unsigned)(unsigned long long)(const void*)sF;  // addr[31:0] = LDS byte offset
        u32x4 g0 = { 1u,                                   // count = 1 valid descriptor
                     ldsAddr,                              // lds_addr [63:32]
                     (unsigned)(gaddr & 0xFFFFFFFFu),      // global_addr lo
                     ((unsigned)(gaddr >> 32) & 0x01FFFFFFu) | 0x80000000u }; // hi | type=2
        i32x8 g1 = { (int)0x07520000,                      // data_size=4B, pad_en, ival=64dw, amt=4dw
                     (int)(64u << 16),                     // tensor_dim0 = 64 (row length)
                     (int)(((unsigned)N & 0xFFFFu) << 16), // tensor_dim1 lo (rows = N -> OOB zero)
                     (int)((((unsigned)N >> 16) & 0xFFFFu) | (64u << 16)), // dim1 hi | tile_dim0=64
                     (int)TM,                              // tile_dim1 = 128, tile_dim2 = 0
                     64,                                   // tensor_dim0_stride = 64 elements
                     0, 0 };                               // dim1_stride unused (2D)
        i32x4 gz = { 0, 0, 0, 0 };
#if defined(__clang_major__) && (__clang_major__ >= 23)
        i32x8 gz8 = { 0, 0, 0, 0, 0, 0, 0, 0 };
        __builtin_amdgcn_tensor_load_to_lds(g0, g1, gz, gz, gz8, 0);
#else
        __builtin_amdgcn_tensor_load_to_lds(g0, g1, gz, gz, 0);
#endif
    }

    if ((tid & 63) == 1) {                       // warm L2 for the next block's tile
        size_t np = (size_t)min(p0 + TM, N - 1) * CFEAT;
        __builtin_prefetch(feats + np, 0, 1);
    }

    // ---- stage weights (transposed) + biases while the TDM runs ----
    for (int idx = tid; idx < CFEAT * CFEAT; idx += 256) {
        int k = idx >> 6, n = idx & 63;
        sW1t[n * CFEAT + k] = (__bf16)W1[idx];
        sW2t[n * CFEAT + k] = (__bf16)W2[idx];
    }
    if (tid < CFEAT) { sB1[tid] = b1[tid]; sB2[tid] = b2[tid]; }
    if (tid < TM) {
        int p = p0 + tid;
        unsigned s = 0; float ic = 0.0f;
        if (p < N) { s = best[p]; ic = 1.0f / fmaxf(hcnt[s], 1.0f); }
        sSlot[tid] = s; sInvC[tid] = ic;
    }
    if (tid == 0) __builtin_amdgcn_s_wait_tensorcnt(0);   // TDM tile landed in sF
    __syncthreads();

    // ---- residual tile: res = feats - voxmean  (feats read from LDS) ----
    for (int idx = tid; idx < TM * CFEAT; idx += 256) {
        int m = idx >> 6, c = idx & 63;
        int p = p0 + m;
        float fv = sF[m * FPITCH + c];            // zero for OOB rows (TDM)
        float rv = 0.0f;
        if (p < N) rv = fv - hsum[(size_t)sSlot[m] * CFEAT + c] * sInvC[m];
        sA[m * APITCH + c] = (__bf16)rv;
    }
    __syncthreads();

    const int lane = tid & 31;
    const int wv   = tid >> 5;
    const int mrow = wv * 16;
    const int ncol = lane & 15;
    const int rb   = (lane >> 4) << 3;          // D-matrix row base (0 or 8)

    // ---- layer 1: h = relu(res@W1 + b1) * feats -> sH (bf16) ----
    #pragma unroll
    for (int nt = 0; nt < 4; ++nt) {
        v8f acc = {};
        #pragma unroll
        for (int ks = 0; ks < 2; ++ks) {
            v16bf a  = load_frag(sA,   APITCH, mrow,    ks * 32, lane);
            v16bf bb = load_frag(sW1t, CFEAT,  nt * 16, ks * 32, lane);
            acc = __builtin_amdgcn_wmma_f32_16x16x32_bf16(
                      false, a, false, bb, (short)0, acc, false, false);
        }
        int n = nt * 16 + ncol;
        float bias = sB1[n];
        #pragma unroll
        for (int j = 0; j < 8; ++j) {
            int m = mrow + rb + j;
            float h = fmaxf(acc[j] + bias, 0.0f) * sF[m * FPITCH + n];
            sH[m * APITCH + n] = (__bf16)h;
        }
    }
    __syncthreads();

    // ---- layer 2: out = relu(h@W2 + b2) -> msOut ----
    #pragma unroll
    for (int nt = 0; nt < 4; ++nt) {
        v8f acc = {};
        #pragma unroll
        for (int ks = 0; ks < 2; ++ks) {
            v16bf a  = load_frag(sH,   APITCH, mrow,    ks * 32, lane);
            v16bf bb = load_frag(sW2t, CFEAT,  nt * 16, ks * 32, lane);
            acc = __builtin_amdgcn_wmma_f32_16x16x32_bf16(
                      false, a, false, bb, (short)0, acc, false, false);
        }
        int n = nt * 16 + ncol;
        float bias = sB2[n];
        #pragma unroll
        for (int j = 0; j < 8; ++j) {
            int m = mrow + rb + j;
            int p = p0 + m;
            if (p < N) msOut[(size_t)p * CFEAT + n] = fmaxf(acc[j] + bias, 0.0f);
        }
    }
}

// ---------------------------------------------------------------------------
// Phase 5: attention mix:  att = sigmoid((Σ_s ms_s) @ Wa^T + ba),
//                          out = Σ_s ms_s * att_s
// ---------------------------------------------------------------------------
__global__ __launch_bounds__(256) void k_final(const float* __restrict__ ms,
                                               const float* __restrict__ Wa,
                                               const float* __restrict__ ba,
                                               float* __restrict__ out, int N) {
    __shared__ float sWa[3 * CFEAT];
    __shared__ float sBa[3];
    int tid = threadIdx.x;
    if (tid < 3 * CFEAT) sWa[tid] = Wa[tid];
    if (tid < 3) sBa[tid] = ba[tid];
    __syncthreads();

    int p = blockIdx.x * blockDim.x + tid;
    if (p >= N) return;

    float dot0 = 0.f, dot1 = 0.f, dot2 = 0.f;
    #pragma unroll
    for (int s = 0; s < 3; ++s) {
        const float4* row = (const float4*)(ms + ((size_t)s * N + p) * CFEAT);
        #pragma unroll
        for (int q = 0; q < CFEAT / 4; ++q) {
            float4 v = row[q];
            int c = q * 4;
            dot0 += v.x * sWa[0 * CFEAT + c] + v.y * sWa[0 * CFEAT + c + 1]
                  + v.z * sWa[0 * CFEAT + c + 2] + v.w * sWa[0 * CFEAT + c + 3];
            dot1 += v.x * sWa[1 * CFEAT + c] + v.y * sWa[1 * CFEAT + c + 1]
                  + v.z * sWa[1 * CFEAT + c + 2] + v.w * sWa[1 * CFEAT + c + 3];
            dot2 += v.x * sWa[2 * CFEAT + c] + v.y * sWa[2 * CFEAT + c + 1]
                  + v.z * sWa[2 * CFEAT + c + 2] + v.w * sWa[2 * CFEAT + c + 3];
        }
    }
    float att0 = 1.0f / (1.0f + expf(-(dot0 + sBa[0])));
    float att1 = 1.0f / (1.0f + expf(-(dot1 + sBa[1])));
    float att2 = 1.0f / (1.0f + expf(-(dot2 + sBa[2])));

    const float4* r0 = (const float4*)(ms + ((size_t)0 * N + p) * CFEAT);
    const float4* r1 = (const float4*)(ms + ((size_t)1 * N + p) * CFEAT);
    const float4* r2 = (const float4*)(ms + ((size_t)2 * N + p) * CFEAT);
    float4* o = (float4*)(out + (size_t)p * CFEAT);
    #pragma unroll
    for (int q = 0; q < CFEAT / 4; ++q) {
        float4 a = r0[q], b = r1[q], c = r2[q], r;
        r.x = a.x * att0 + b.x * att1 + c.x * att2;
        r.y = a.y * att0 + b.y * att1 + c.y * att2;
        r.z = a.z * att0 + b.z * att1 + c.z * att2;
        r.w = a.w * att0 + b.w * att1 + c.w * att2;
        o[q] = r;
    }
}

// ---------------------------------------------------------------------------
// Host launcher
// ---------------------------------------------------------------------------
extern "C" void kernel_launch(void* const* d_in, const int* in_sizes, int n_in,
                              void* d_out, int out_size, void* d_ws, size_t ws_size,
                              hipStream_t stream) {
    const float* feats  = (const float*)d_in[0];
    const int*   coords = (const int*)  d_in[1];
    const float* W1     = (const float*)d_in[2];
    const float* b1     = (const float*)d_in[3];
    const float* W2     = (const float*)d_in[4];
    const float* b2     = (const float*)d_in[5];
    const float* Wa     = (const float*)d_in[6];
    const float* ba     = (const float*)d_in[7];
    float*       out    = (float*)d_out;

    const int N = in_sizes[0] / CFEAT;

    // Workspace layout (~652 MB): hash keys/cnt/sums, slot maps, per-scale ms.
    char* w = (char*)d_ws;
    unsigned* hKey  = (unsigned*)w;  w += (size_t)HSIZE * 4;
    float*    hCnt  = (float*)w;     w += (size_t)HSIZE * 4;
    float*    hSum  = (float*)w;     w += (size_t)HSIZE * CFEAT * 4;
    unsigned* slots = (unsigned*)w;  w += (size_t)N * 4;
    unsigned* best  = (unsigned*)w;  w += (size_t)N * 4;
    float*    ms    = (float*)w;     // 3 * N * 64 f32

    const int ptBlocks  = (N + 255) / 256;
    const int scBlocks  = (int)(((long long)N * CFEAT + 255) / 256);
    const int mlpBlocks = (N + TM - 1) / TM;
    static const int kScales[3] = {2, 4, 8};

    for (int i = 0; i < 3; ++i) {
        const int scale = kScales[i];
        hipMemsetAsync(hKey, 0xFF, (size_t)HSIZE * 4, stream);
        hipMemsetAsync(hCnt, 0x00, (size_t)HSIZE * 4, stream);
        hipMemsetAsync(hSum, 0x00, (size_t)HSIZE * CFEAT * 4, stream);

        k_insert <<<ptBlocks, 256, 0, stream>>>(coords, hKey, hCnt, slots, N, scale);
        k_scatter<<<scBlocks, 256, 0, stream>>>(feats, slots, hSum, N);
        k_corner <<<ptBlocks, 256, 0, stream>>>(coords, hKey, slots, best, N, scale);
        k_mlp    <<<mlpBlocks, 256, 0, stream>>>(feats, hSum, hCnt, best,
                                                 W1 + (size_t)i * CFEAT * CFEAT,
                                                 b1 + (size_t)i * CFEAT,
                                                 W2 + (size_t)i * CFEAT * CFEAT,
                                                 b2 + (size_t)i * CFEAT,
                                                 ms + (size_t)i * N * CFEAT, N);
    }
    k_final<<<ptBlocks, 256, 0, stream>>>(ms, Wa, ba, out, N);
}
